// TernaryConv2d_64639257805187
// MI455X (gfx1250) — matrix operand order repaired and checked by reading, not compiled, not run on hardware
//
#include <hip/hip_runtime.h>

typedef __bf16 bf16;
typedef __attribute__((ext_vector_type(8)))  __bf16 v8bf;
typedef __attribute__((ext_vector_type(16))) __bf16 v16bf;
typedef __attribute__((ext_vector_type(8)))  float  v8f;

// exact parameter type reported by clang for the async-LDS builtin:
typedef int i32x4 __attribute__((vector_size(4 * sizeof(int))));
typedef __attribute__((address_space(1))) i32x4* as1_i32x4_ptr;
typedef __attribute__((address_space(3))) i32x4* as3_i32x4_ptr;

#define C_IN   64
#define C_OUT  64
#define HH     224
#define WW     224
#define NBATCH 16
#define TILE_W 128
#define HALO_W (TILE_W + 2)                  // 130
#define ROWSKEW 72                           // 64 + 8 bf16: 36-dword stride => conflict-free banks
#define W_GLB_BYTES (9 * C_OUT * C_IN * 2)   // 73728 B packed tern weights in global ws
#define W_LDS_BYTES (9 * C_OUT * ROWSKEW * 2)        // 82944 B (skewed rows)
#define X_LDS_BYTES (3 * HALO_W * ROWSKEW * 2)       // 56160 B (skewed rows)
#define LDS_BYTES (W_LDS_BYTES + X_LDS_BYTES)        // 139104 B

// ---------------------------------------------------------------------------
// Kernel 1: TWN ternarization + pack into [tap][c_out][c_in] bf16 + alpha[c].
// ---------------------------------------------------------------------------
__global__ void ternarize_pack_kernel(const float* __restrict__ w,
                                      bf16* __restrict__ wt,
                                      float* __restrict__ alpha_out) {
  int c = blockIdx.x * blockDim.x + threadIdx.x;
  if (c >= C_OUT) return;
  const float* wc = w + c * (C_IN * 9);
  float s = 0.f;
  for (int i = 0; i < C_IN * 9; ++i) s += fabsf(wc[i]);
  float delta = (0.7f / (float)(C_IN * 9)) * s;
  float asum = 0.f;
  int cnt = 0;
  for (int i = 0; i < C_IN * 9; ++i) {
    float a = fabsf(wc[i]);
    if (a > delta) { asum += a; ++cnt; }
  }
  if (cnt == 0) cnt = 1;
  float alpha = fmaxf(asum / (float)cnt, 1e-4f);
  alpha_out[c] = alpha;
  for (int ci = 0; ci < C_IN; ++ci) {
    for (int t = 0; t < 9; ++t) {  // t = kh*3+kw
      float v = wc[ci * 9 + t];
      float tv = (v > delta) ? 1.f : ((v < -delta) ? -1.f : 0.f);
      wt[((size_t)t * C_OUT + c) * C_IN + ci] = (bf16)tv;
    }
  }
}

// ---------------------------------------------------------------------------
// Kernel 2: implicit-GEMM conv via v_wmma_f32_16x16x32_bf16.
//   Block = 256 threads (8 waves), computes 128 pixels x 64 channels of one
//   (n, h) row. A = tern weights (M = c_out), B = x pixels (N = w), so f32
//   C-tile stores are contiguous along w. LDS rows skewed to 72 bf16 so the
//   per-lane 144 B stride maps 16 lanes onto 16 distinct bank offsets.
// ---------------------------------------------------------------------------
__global__ __launch_bounds__(256)
void tern_conv_wmma_kernel(const float* __restrict__ x,
                           const bf16* __restrict__ wtg,
                           const float* __restrict__ alpha,
                           const float* __restrict__ bias,
                           float* __restrict__ out) {
  extern __shared__ char smem[];
  bf16* wt = (bf16*)smem;                       // [9][C_OUT][ROWSKEW]
  bf16* xs = (bf16*)(smem + W_LDS_BYTES);       // [3][HALO_W][ROWSKEW]

  const int tid  = threadIdx.x;
  const int lane = tid & 31;
  const int wave = tid >> 5;
  const int hi   = lane >> 4;   // half-wave select
  const int ln   = lane & 15;

  const int b    = blockIdx.x;
  const int tile = b & 1;
  const int h    = (b >> 1) % HH;
  const int n    = b / (2 * HH);
  const int w0   = tile * TILE_W;

  // ---- stage ternarized weights global->LDS (async-LDS path on CDNA5) ----
  // global rows are 128 B (64 bf16); LDS rows are 144 B; 8 x 16 B chunks/row.
  const int NCHUNK = W_GLB_BYTES / 16;  // 4608
#if defined(__AMDGCN__) && __has_builtin(__builtin_amdgcn_global_load_async_to_lds_b128)
  for (int cidx = tid; cidx < NCHUNK; cidx += 256) {
    int row = cidx >> 3;
    int col = (cidx & 7) * 16;
    __builtin_amdgcn_global_load_async_to_lds_b128(
        (as1_i32x4_ptr)((char*)wtg + cidx * 16),
        (as3_i32x4_ptr)((char*)wt + row * (ROWSKEW * 2) + col),
        /*offset=*/0, /*cpol=*/0);
  }
#else
  for (int cidx = tid; cidx < NCHUNK; cidx += 256) {
    int row = cidx >> 3;
    int col = (cidx & 7) * 16;
    *(uint4*)((char*)wt + row * (ROWSKEW * 2) + col) =
        *(const uint4*)((const char*)wtg + cidx * 16);
  }
#endif

  // ---- stage x halo (3 rows x 130 w x 64 c), f32 -> bf16, layout [r][w][c]
  const int XTOT = 3 * C_IN * HALO_W;
  for (int idx = tid; idx < XTOT; idx += 256) {
    int wi = idx % HALO_W;            // fastest => coalesced global reads
    int rc = idx / HALO_W;
    int c  = rc & (C_IN - 1);
    int r  = rc >> 6;
    int gw = w0 + wi - 1;
    int gh = h + r - 1;
    float v = 0.f;
    if ((unsigned)gw < WW && (unsigned)gh < HH)
      v = x[(((size_t)n * C_IN + c) * HH + gh) * WW + gw];
    xs[(size_t)(r * HALO_W + wi) * ROWSKEW + c] = (bf16)v;
  }

#if defined(__AMDGCN__) && __has_builtin(__builtin_amdgcn_global_load_async_to_lds_b128)
#if __has_builtin(__builtin_amdgcn_s_wait_asynccnt)
  __builtin_amdgcn_s_wait_asynccnt(0);
#else
  asm volatile("s_wait_asynccnt 0" ::: "memory");
#endif
#endif
  __syncthreads();

  // ---- wave tiling: cgrp = 16 output channels, mhalf = 64 pixels ----
  const int cgrp  = wave & 3;
  const int mhalf = wave >> 2;
  const int cbase = cgrp * 16;
  const int arow  = cbase + ln;   // A-matrix row (c_out) held by this lane

  v8f acc[4] = {};   // 4 pixel-subtiles of 16

#pragma unroll
  for (int tap = 0; tap < 9; ++tap) {
    const int dh = tap / 3;
    const int dw = tap % 3;
    const bf16* wtap = wt + (size_t)(tap * C_OUT + arow) * ROWSKEW;
    const bf16* xrow = xs + (size_t)(dh * HALO_W) * ROWSKEW;
#pragma unroll
    for (int kc = 0; kc < 2; ++kc) {
      // A fragment (16x32 bf16): per lane K = {8*hi+0..7, 16+8*hi+0..7}
      const bf16* ap = wtap + kc * 32 + hi * 8;
      v8bf alo = *(const v8bf*)ap;
      v8bf ahi = *(const v8bf*)(ap + 16);
      v16bf a = __builtin_shufflevector(alo, ahi,
          0, 1, 2, 3, 4, 5, 6, 7, 8, 9, 10, 11, 12, 13, 14, 15);
      const int kb = kc * 32 + hi * 8;  // B per-lane K = {8*hi+0..7, 16+8*hi+0..7}
#pragma unroll
      for (int msub = 0; msub < 4; ++msub) {
        const int wpix = mhalf * 64 + msub * 16 + ln + dw;  // LDS w index
        const bf16* bp = xrow + (size_t)wpix * ROWSKEW + kb;
        v8bf blo = *(const v8bf*)bp;
        v8bf bhi = *(const v8bf*)(bp + 16);
        v16bf bm = __builtin_shufflevector(blo, bhi,
            0, 1, 2, 3, 4, 5, 6, 7, 8, 9, 10, 11, 12, 13, 14, 15);
        acc[msub] = __builtin_amdgcn_wmma_f32_16x16x32_bf16(
            false, a, false, bm, (short)0, acc[msub], false, false);
      }
    }
  }

  // ---- epilogue: out = acc * alpha[c] + bias[c], coalesced along w ----
  float av[8], bv[8];
#pragma unroll
  for (int r = 0; r < 8; ++r) {
    int cc = cbase + hi * 8 + r;   // C layout: VGPR r, lanes16-31 => M = r+8
    av[r] = alpha[cc];
    bv[r] = bias[cc];
  }
#pragma unroll
  for (int msub = 0; msub < 4; ++msub) {
    int wp = w0 + mhalf * 64 + msub * 16 + ln;
    if (wp < WW) {
#pragma unroll
      for (int r = 0; r < 8; ++r) {
        int cc = cbase + hi * 8 + r;
        out[(((size_t)n * C_OUT + cc) * HH + h) * WW + wp] =
            acc[msub][r] * av[r] + bv[r];
      }
    }
  }
}

// ---------------------------------------------------------------------------
extern "C" void kernel_launch(void* const* d_in, const int* in_sizes, int n_in,
                              void* d_out, int out_size, void* d_ws, size_t ws_size,
                              hipStream_t stream) {
  const float* x      = (const float*)d_in[0];
  const float* weight = (const float*)d_in[1];
  const float* bias   = (const float*)d_in[2];
  float* out = (float*)d_out;

  bf16*  wt    = (bf16*)d_ws;                          // 73728 B packed
  float* alpha = (float*)((char*)d_ws + W_GLB_BYTES);  // 64 floats

  ternarize_pack_kernel<<<1, 64, 0, stream>>>(weight, wt, alpha);

  (void)hipFuncSetAttribute((const void*)tern_conv_wmma_kernel,
                            hipFuncAttributeMaxDynamicSharedMemorySize,
                            LDS_BYTES);
  const int blocks = NBATCH * HH * 2;  // 7168
  tern_conv_wmma_kernel<<<blocks, 256, LDS_BYTES, stream>>>(
      x, wt, alpha, bias, out);
}
